// CausalSelfAttention_58317065945297
// MI455X (gfx1250) — matrix operand order repaired and checked by reading, not compiled, not run on hardware
//
#include <hip/hip_runtime.h>
#include <hip/hip_bf16.h>

typedef __attribute__((ext_vector_type(16))) _Float16 v16h;
typedef __attribute__((ext_vector_type(8)))  _Float16 v8h;
typedef __attribute__((ext_vector_type(8)))  float    v8f;
typedef __attribute__((ext_vector_type(4)))  float    v4f;

#define HCNT 16
#define HD   64
#define DMODEL 1024
#define LDS_PITCH 72   // halves; 144B row pitch: 16B aligned, bank-staggered

// ---------------------------------------------------------------------------
// WMMA helper: D = A(16x32 f16) * B(32x16 f16) + C(16x16 f32)
// ---------------------------------------------------------------------------
static __device__ __forceinline__ v8f wmma_f16(v16h a, v16h b, v8f c) {
  return __builtin_amdgcn_wmma_f32_16x16x32_f16(
      /*neg_a=*/false, a, /*neg_b=*/false, b,
      /*c_mod=*/(short)0, c, /*reuse_a=*/false, /*reuse_b=*/false);
}

// Async 16-byte global -> LDS copy (per-lane), tracked by ASYNCcnt.
static __device__ __forceinline__ void async_ld16(unsigned lds_off, const void* gptr) {
  asm volatile("global_load_async_to_lds_b128 %0, %1, off"
               :: "v"(lds_off), "v"(gptr) : "memory");
}

template <int N>
static __device__ __forceinline__ void wait_asynccnt() {
#if __has_builtin(__builtin_amdgcn_s_wait_asynccnt)
  __builtin_amdgcn_s_wait_asynccnt(N);
#else
  asm volatile("s_wait_asynccnt %0" :: "i"(N) : "memory");
#endif
}

// A-matrix fragment (16x32, f16) from row-major storage (global or LDS).
// ISA layout: lanes 0-15 -> M=lane, K in {0..7, 16..23}; lanes 16-31 -> K in
// {8..15, 24..31}.  Two contiguous 16-byte loads per lane.
static __device__ __forceinline__ v16h load_a_f16(const _Float16* base,
                                                  int ld, int row0, int kb, int lane) {
  const int r    = lane & 15;
  const int koff = (lane >> 4) << 3;           // 0 or 8
  const _Float16* p = base + (size_t)(row0 + r) * ld + kb + koff;
  v8h lo = *(const v8h*)p;
  v8h hi = *(const v8h*)(p + 16);
  v16h a;
#pragma unroll
  for (int i = 0; i < 8; ++i) { a[i] = lo[i]; a[i + 8] = hi[i]; }
  return a;
}

// B-matrix fragment (32x16, f16) from Bt stored N x K row-major.
// ISA layout: lanes 0-15 -> N=lane, K in {0..7, 16..23}; lanes 16-31 -> K in
// {8..15, 24..31}.
static __device__ __forceinline__ v16h load_b_f16(const _Float16* bt,
                                                  int ld, int n0, int kb, int lane) {
  const int n    = lane & 15;
  const int koff = (lane >> 4) << 3;           // 0 or 8
  const _Float16* p = bt + (size_t)(n0 + n) * ld + kb + koff;
  v8h lo = *(const v8h*)p;
  v8h hi = *(const v8h*)(p + 16);
  v16h b;
#pragma unroll
  for (int i = 0; i < 8; ++i) { b[i] = lo[i]; b[i + 8] = hi[i]; }
  return b;
}

// A-matrix fragment from an f32 row-major source, converted to f16.
static __device__ __forceinline__ v16h load_a_f32cvt(const float* __restrict__ base,
                                                     int ld, int row0, int kb, int lane) {
  const int r    = lane & 15;
  const int koff = (lane >> 4) << 3;
  const float* p = base + (size_t)(row0 + r) * ld + kb + koff;
  v4f c0 = *(const v4f*)(p);
  v4f c1 = *(const v4f*)(p + 4);
  v4f c2 = *(const v4f*)(p + 16);
  v4f c3 = *(const v4f*)(p + 20);
  v16h a;
#pragma unroll
  for (int i = 0; i < 4; ++i) {
    a[i]      = (_Float16)c0[i];
    a[i + 4]  = (_Float16)c1[i];
    a[i + 8]  = (_Float16)c2[i];
    a[i + 12] = (_Float16)c3[i];
  }
  return a;
}

// ---------------------------------------------------------------------------
// 1) f32 -> f16 convert
// ---------------------------------------------------------------------------
__global__ void cvt_f32_to_f16(const float* __restrict__ src,
                               _Float16* __restrict__ dst, int n) {
  int i = blockIdx.x * blockDim.x + threadIdx.x;
  if (i < n) dst[i] = (_Float16)src[i];
}

// ---------------------------------------------------------------------------
// 2) GEMM  Y = A(MxK f16) * Bt(NxK f16)^T + bias
//    256 threads = 8 waves (4x2); wave tile 32x32; WG tile 128x64.
//    K is consumed in 64-wide chunks staged in LDS by async global->LDS
//    copies, double-buffered so DMA of chunk i+1 overlaps WMMA on chunk i.
//    mode 0: f16 out, (B,H,S,Hd) head layout   (Q, K projections)
//    mode 1: f16 out, (B,H,Hd,S) transposed    (V projection -> Vt)
//    mode 2: f32 out, row-major (M,N)          (output projection)
// ---------------------------------------------------------------------------
__global__ __launch_bounds__(256) void gemm_xwT(
    const _Float16* __restrict__ A, const _Float16* __restrict__ Bt,
    const float* __restrict__ bias, _Float16* __restrict__ out16,
    float* __restrict__ out32, int M, int N, int K, int S, int mode) {
  __shared__ __align__(16) _Float16 smemA[2][128][LDS_PITCH];
  __shared__ __align__(16) _Float16 smemB[2][64][LDS_PITCH];

  const int lane = threadIdx.x & 31;
  const int wv   = threadIdx.x >> 5;
  const int wr   = wv & 3, wc = wv >> 2;        // 4x2 wave grid
  const int tiles_n = N / 64;
  const int bm = blockIdx.x / tiles_n, bn = blockIdx.x % tiles_n;
  const int row_wg = bm * 128;
  const int col_wg = bn * 64;

  // Issue one 64-wide K chunk: A tile 128x64 (4 copies/thread),
  // B tile 64x64 (2 copies/thread).  6 async ops per thread per chunk.
  auto issue_chunk = [&](int kb, int buf) {
    const unsigned aBase = (unsigned)(size_t)&smemA[buf][0][0];
    const unsigned bBase = (unsigned)(size_t)&smemB[buf][0][0];
#pragma unroll
    for (int j = 0; j < 4; ++j) {
      const int idx = threadIdx.x + 256 * j;          // 1024 chunks of 16B
      const int r = idx >> 3, c = idx & 7;
      const _Float16* g = A + (size_t)(row_wg + r) * K + kb + c * 8;
      async_ld16(aBase + (unsigned)(r * (LDS_PITCH * 2) + c * 16), g);
    }
#pragma unroll
    for (int j = 0; j < 2; ++j) {
      const int idx = threadIdx.x + 256 * j;          // 512 chunks of 16B
      const int r = idx >> 3, c = idx & 7;
      const _Float16* g = Bt + (size_t)(col_wg + r) * K + kb + c * 8;
      async_ld16(bBase + (unsigned)(r * (LDS_PITCH * 2) + c * 16), g);
    }
  };

  v8f acc00 = {}, acc01 = {}, acc10 = {}, acc11 = {};
  const int nc = K / 64;

  issue_chunk(0, 0);
  for (int ic = 0; ic < nc; ++ic) {
    const int cur = ic & 1;
    if (ic + 1 < nc) {
      issue_chunk((ic + 1) * 64, cur ^ 1);
      wait_asynccnt<6>();            // chunk ic landed; ic+1 still in flight
    } else {
      wait_asynccnt<0>();
    }
    __syncthreads();

    const _Float16* Ab = &smemA[cur][0][0];
    const _Float16* Bb = &smemB[cur][0][0];
#pragma unroll
    for (int kk = 0; kk < 64; kk += 32) {
      v16h a0 = load_a_f16(Ab, LDS_PITCH, wr * 32,      kk, lane);
      v16h a1 = load_a_f16(Ab, LDS_PITCH, wr * 32 + 16, kk, lane);
      v16h b0 = load_b_f16(Bb, LDS_PITCH, wc * 32,      kk, lane);
      v16h b1 = load_b_f16(Bb, LDS_PITCH, wc * 32 + 16, kk, lane);
      acc00 = wmma_f16(a0, b0, acc00);
      acc01 = wmma_f16(a0, b1, acc01);
      acc10 = wmma_f16(a1, b0, acc10);
      acc11 = wmma_f16(a1, b1, acc11);
    }
    __syncthreads();                 // reads done before buf is overwritten
  }

  const int row0 = row_wg + wr * 32;
  const int col0 = col_wg + wc * 32;
  const int nloc  = lane & 15;
  const int rbase = (lane >> 4) << 3;
  v8f accs[4] = {acc00, acc01, acc10, acc11};
#pragma unroll
  for (int t = 0; t < 4; ++t) {
    const int ti = t >> 1, tj = t & 1;
    const int n  = col0 + tj * 16 + nloc;
    const float bv = bias[n];
#pragma unroll
    for (int v = 0; v < 8; ++v) {
      const int m = row0 + ti * 16 + rbase + v;
      const float val = accs[t][v] + bv;
      if (mode == 2) {
        out32[(size_t)m * N + n] = val;
      } else {
        const int b = m / S, s = m % S;
        const int h = n >> 6, nd = n & 63;
        const size_t idx = (mode == 0)
            ? (((size_t)(b * HCNT + h) * S + s) * HD + nd)     // (B,H,S,Hd)
            : (((size_t)(b * HCNT + h) * HD + nd) * S + s);    // (B,H,Hd,S)
        out16[idx] = (_Float16)val;
      }
    }
  }
}

// ---------------------------------------------------------------------------
// 3) Scores + causal softmax.  4 waves/WG; wave owns 16 query rows.
//    Pass 1: stream key tiles (2 WMMAs each, K=Hd=64), write raw scaled
//    scores, track online row max / sum-of-exp in registers.
//    Pass 2: normalize in place; exact 0.0 above the diagonal.
// ---------------------------------------------------------------------------
__global__ __launch_bounds__(128) void attn_scores_softmax(
    const _Float16* __restrict__ Q, const _Float16* __restrict__ Km,
    float* __restrict__ attn, int S, float scale) {
  const int lane = threadIdx.x & 31;
  const int wv   = threadIdx.x >> 5;
  const int q0   = blockIdx.x * 64 + wv * 16;
  const _Float16* Qh = Q  + (size_t)blockIdx.y * S * HD;
  const _Float16* Kh = Km + (size_t)blockIdx.y * S * HD;
  float* Ah          = attn + (size_t)blockIdx.y * S * S;

  const v16h qa0 = load_a_f16(Qh, HD, q0, 0,  lane);
  const v16h qa1 = load_a_f16(Qh, HD, q0, 32, lane);

  const int nloc  = lane & 15;
  const int rbase = (lane >> 4) << 3;
  const float NEG_INF = -__builtin_inff();
  float mrow[8], lrow[8];
#pragma unroll
  for (int v = 0; v < 8; ++v) { mrow[v] = NEG_INF; lrow[v] = 0.0f; }

  const int ktmax = (q0 + 15) >> 4;             // inclusive causal limit
  for (int kt = 0; kt <= ktmax; ++kt) {
    const int k0 = kt * 16;
    v16h kb0 = load_b_f16(Kh, HD, k0, 0,  lane);
    v16h kb1 = load_b_f16(Kh, HD, k0, 32, lane);
    v8f sc = {};
    sc = wmma_f16(qa0, kb0, sc);
    sc = wmma_f16(qa1, kb1, sc);

    const int col = k0 + nloc;
    float sv[8];
#pragma unroll
    for (int v = 0; v < 8; ++v) {
      const int row = q0 + rbase + v;
      const float s = sc[v] * scale;
      const bool ok = (col <= row);
      sv[v] = ok ? s : NEG_INF;
      Ah[(size_t)row * S + col] = ok ? s : 0.0f;  // raw; pass 2 re-masks
    }
    // Online softmax: reduce within each 16-lane half (rows live per-half).
#pragma unroll
    for (int v = 0; v < 8; ++v) {
      float t = sv[v];
#pragma unroll
      for (int m = 1; m <= 8; m <<= 1) t = fmaxf(t, __shfl_xor(t, m, 32));
      const float mn = fmaxf(mrow[v], t);
      float p = __expf(sv[v] - mn);
#pragma unroll
      for (int m = 1; m <= 8; m <<= 1) p += __shfl_xor(p, m, 32);
      lrow[v] = lrow[v] * __expf(mrow[v] - mn) + p;
      mrow[v] = mn;
    }
  }

  // Pass 2: normalize own 16 rows (same-wave store->load is ordered).
  for (int r = 0; r < 16; ++r) {
    const int row  = q0 + r;
    const int src  = (r < 8) ? 0 : 16;
    const float mr = __shfl(mrow[r & 7], src, 32);
    const float lr = __shfl(lrow[r & 7], src, 32);
    const float inv = 1.0f / lr;
    float* rowp = Ah + (size_t)row * S;
    for (int c0 = lane * 4; c0 < S; c0 += 128) {
      v4f x = *(const v4f*)(rowp + c0);
      v4f y;
#pragma unroll
      for (int i = 0; i < 4; ++i) {
        const int c = c0 + i;
        y[i] = (c <= row) ? __expf(x[i] - mr) * inv : 0.0f;
      }
      *(v4f*)(rowp + c0) = y;
    }
  }
}

// ---------------------------------------------------------------------------
// 4) ctx = P @ V.  Wave owns 16 rows x full Hd=64 (4 accumulators).
//    A = attn f32 (from d_out) converted to f16; Bt = Vt (B,H,Hd,S).
//    Causal: only iterate keys up to the diagonal (zeros beyond).
// ---------------------------------------------------------------------------
__global__ __launch_bounds__(128) void attn_ctx(
    const float* __restrict__ attn, const _Float16* __restrict__ Vt,
    _Float16* __restrict__ ctx16, int S) {
  const int lane = threadIdx.x & 31;
  const int wv   = threadIdx.x >> 5;
  const int bh   = blockIdx.y;
  const int b    = bh >> 4, h = bh & 15;
  const int q0   = blockIdx.x * 64 + wv * 16;
  const float* Ah     = attn + (size_t)bh * S * S;
  const _Float16* Vh  = Vt   + (size_t)bh * HD * S;

  v8f acc[4] = {};
  const int kend = q0 + 16;                      // valid keys for this block
  for (int kb = 0; kb < kend; kb += 32) {
    v16h a = load_a_f32cvt(Ah, S, q0, kb, lane); // zeros past diag are exact
#pragma unroll
    for (int t = 0; t < 4; ++t) {
      v16h bf = load_b_f16(Vh, S, t * 16, kb, lane);
      acc[t] = wmma_f16(a, bf, acc[t]);
    }
  }

  const int nloc  = lane & 15;
  const int rbase = (lane >> 4) << 3;
#pragma unroll
  for (int t = 0; t < 4; ++t) {
    const int nd = t * 16 + nloc;
#pragma unroll
    for (int v = 0; v < 8; ++v) {
      const int s = q0 + rbase + v;
      ctx16[((size_t)(b * S + s)) * DMODEL + h * HD + nd] = (_Float16)acc[t][v];
    }
  }
}

// ---------------------------------------------------------------------------
extern "C" void kernel_launch(void* const* d_in, const int* in_sizes, int n_in,
                              void* d_out, int out_size, void* d_ws, size_t ws_size,
                              hipStream_t stream) {
  (void)in_sizes; (void)n_in; (void)out_size; (void)ws_size;
  const float* x  = (const float*)d_in[0];
  const float* Wq = (const float*)d_in[1];
  const float* bq = (const float*)d_in[2];
  const float* Wk = (const float*)d_in[3];
  const float* bk = (const float*)d_in[4];
  const float* Wv = (const float*)d_in[5];
  const float* bv = (const float*)d_in[6];
  const float* Wo = (const float*)d_in[7];
  const float* bo = (const float*)d_in[8];

  const int B = 2, S = 2048, D = DMODEL, BH = B * HCNT, M = B * S;
  float* out  = (float*)d_out;                 // (B,S,D)
  float* attn = out + (size_t)M * D;           // (B,H,S,S)

  size_t off = 0;
  auto alloc16 = [&](size_t elems) {
    _Float16* p = (_Float16*)((char*)d_ws + off);
    off += elems * sizeof(_Float16);
    return p;
  };
  _Float16* x16   = alloc16((size_t)M * D);
  _Float16* wq16  = alloc16((size_t)D * D);
  _Float16* wk16  = alloc16((size_t)D * D);
  _Float16* wv16  = alloc16((size_t)D * D);
  _Float16* wo16  = alloc16((size_t)D * D);
  _Float16* q16   = alloc16((size_t)M * D);    // (B,H,S,Hd)
  _Float16* k16   = alloc16((size_t)M * D);    // (B,H,S,Hd)
  _Float16* vt16  = alloc16((size_t)M * D);    // (B,H,Hd,S)
  _Float16* ctx16 = alloc16((size_t)M * D);    // (B*S, D)
  // total workspace: 48 MB

  // Stage 1: precision convert.
  cvt_f32_to_f16<<<(M * D + 255) / 256, 256, 0, stream>>>(x, x16, M * D);
  cvt_f32_to_f16<<<(D * D + 255) / 256, 256, 0, stream>>>(Wq, wq16, D * D);
  cvt_f32_to_f16<<<(D * D + 255) / 256, 256, 0, stream>>>(Wk, wk16, D * D);
  cvt_f32_to_f16<<<(D * D + 255) / 256, 256, 0, stream>>>(Wv, wv16, D * D);
  cvt_f32_to_f16<<<(D * D + 255) / 256, 256, 0, stream>>>(Wo, wo16, D * D);

  // Stage 2: Q/K/V projections (async-LDS staged WMMA GEMM).
  const int gemm_blocks = (M / 128) * (D / 64);
  gemm_xwT<<<gemm_blocks, 256, 0, stream>>>(x16, wq16, bq, q16, nullptr, M, D, D, S, 0);
  gemm_xwT<<<gemm_blocks, 256, 0, stream>>>(x16, wk16, bk, k16, nullptr, M, D, D, S, 0);
  gemm_xwT<<<gemm_blocks, 256, 0, stream>>>(x16, wv16, bv, vt16, nullptr, M, D, D, S, 1);

  // Stage 3: causal scores + softmax (attn probabilities -> d_out).
  const float scale = 0.125f;                  // 1/sqrt(64)
  attn_scores_softmax<<<dim3(S / 64, BH), 128, 0, stream>>>(q16, k16, attn, S, scale);

  // Stage 4: ctx = P @ V.
  attn_ctx<<<dim3(S / 64, BH), 128, 0, stream>>>(attn, vt16, ctx16, S);

  // Stage 5: output projection -> d_out.
  gemm_xwT<<<gemm_blocks, 256, 0, stream>>>(ctx16, wo16, bo, nullptr, out, M, D, D, S, 2);
}
